// AirspaceModel_80444737454217
// MI455X (gfx1250) — matrix operand order repaired and checked by reading, not compiled, not run on hardware
//
#include <hip/hip_runtime.h>

#define DEV_INLINE __device__ __forceinline__

typedef __attribute__((ext_vector_type(16))) __bf16 v16bf;
typedef __attribute__((ext_vector_type(8)))  float  v8f;

namespace {
constexpr int NN  = 126;   // nodes
constexpr int T   = 24;    // seq len
constexpr int F   = 17;    // input features
constexpr int H   = 100;   // hidden
constexpr int B   = 64;    // batch
constexpr int HP  = 112;   // hidden padded to 7*16
constexpr int GP  = 4 * HP;  // 448 padded gate columns (28 N-tiles)
constexpr int KX  = 32;    // input K padded (one bf16 WMMA K step)
constexpr int KH  = 128;   // hidden K padded (four K steps)
constexpr int G4H = 4 * H; // 400 real gate rows
constexpr int NW  = 8;     // waves per block (256 threads, wave32)
}

// ---------- helpers ----------

DEV_INLINE unsigned short f2bf(float f) {
  unsigned int u = __float_as_uint(f);
  unsigned int r = (u + 0x7FFFu + ((u >> 16) & 1u)) >> 16;  // RNE
  return (unsigned short)r;
}

DEV_INLINE float sigf(float x) { return 1.0f / (1.0f + __expf(-x)); }

union BF16Frag { uint4 u[2]; v16bf v; };

// A fragment: 16x32 bf16, row-major source with row stride `stride` elements.
// ISA layout: lanes 0-15 hold K{0..7,16..23}, lanes 16-31 hold K{8..15,24..31}.
DEV_INLINE v16bf load_a_frag(const unsigned short* buf, int mbase, int kbase,
                             int stride, int lane) {
  int m    = mbase + (lane & 15);
  int half = lane >> 4;
  const unsigned short* p = buf + (size_t)m * stride + kbase + half * 8;
  BF16Frag f;
  f.u[0] = *(const uint4*)(p);        // K = kbase + 8*half .. +7
  f.u[1] = *(const uint4*)(p + 16);   // K = kbase + 16 + 8*half .. +7
  return f.v;
}

// B fragment: 32x16 bf16 (K x N), weights stored as W[n][k] row-major,
// row stride `stride` elements.  Lane holds column n = nbase + lane%16,
// lanes 0-15 hold K{0..7,16..23}, lanes 16-31 hold K{8..15,24..31}.
DEV_INLINE v16bf load_b_frag(const unsigned short* W, int nbase, int kbase,
                             int stride, int lane) {
  int ncol = nbase + (lane & 15);
  int half = lane >> 4;
  const unsigned short* p = W + (size_t)ncol * stride + kbase + half * 8;
  BF16Frag f;
  f.u[0] = *(const uint4*)(p);
  f.u[1] = *(const uint4*)(p + 16);
  return f.v;
}

// C/D fragment (16x16 f32): VGPR r, lanes 0-15 -> M=r, lanes 16-31 -> M=r+8.
DEV_INLINE void store_c_frag(float* G, int mbase, int nbase, int lane, v8f c) {
  int ncol = nbase + (lane & 15);
  int mrow = mbase + ((lane >> 4) << 3);
#pragma unroll
  for (int r = 0; r < 8; ++r) G[(size_t)(mrow + r) * GP + ncol] = c[r];
}

DEV_INLINE v8f wmma_bf16(v16bf a, v16bf b, v8f c) {
  return __builtin_amdgcn_wmma_f32_16x16x32_bf16(
      /*neg_a=*/false, a, /*neg_b=*/false, b,
      /*c_mod=*/(short)0, c, /*reuse_a=*/false, /*reuse_b=*/false);
}

// ---------- weight conversion: f32 [N][4H][Kin] -> bf16 [N][GP][KP], zero-padded ----------

__global__ void convert_w_kernel(const float* __restrict__ W,
                                 unsigned short* __restrict__ dst,
                                 int Kin, int KP, int total) {
  for (int idx = blockIdx.x * blockDim.x + threadIdx.x; idx < total;
       idx += gridDim.x * blockDim.x) {
    int k = idx % KP;
    int r = (idx / KP) % GP;
    int n = idx / (KP * GP);
    int gate = r / HP, h = r % HP;
    float v = 0.0f;
    if (h < H && k < Kin)
      v = W[((size_t)n * G4H + gate * H + h) * Kin + k];
    dst[idx] = f2bf(v);
  }
}

// ---------- main fused kernel: one workgroup per node ----------

__global__ __launch_bounds__(256, 1)
void airspace_lstm_kernel(const float* __restrict__ x,
                          const float* __restrict__ b_ih0,
                          const float* __restrict__ b_hh0,
                          const float* __restrict__ b_ih1,
                          const float* __restrict__ b_hh1,
                          const float* __restrict__ w_lin,
                          const float* __restrict__ b_lin,
                          const float* __restrict__ w_end,
                          const float* __restrict__ b_end,
                          const unsigned short* __restrict__ Wih0,
                          const unsigned short* __restrict__ Whh0,
                          const unsigned short* __restrict__ Wih1,
                          const unsigned short* __restrict__ Whh1,
                          float* __restrict__ out,
                          float* __restrict__ hn,
                          float* __restrict__ cn) {
  __shared__ unsigned short Xb[B * KX];     // x_t bf16, padded K
  __shared__ unsigned short H1b[B * KH];    // h1 bf16, padded K
  __shared__ unsigned short H2b[B * KH];    // h2 bf16, padded K
  __shared__ float Gs[B * GP];              // gate pre-activations (f32)
  __shared__ float C1s[B * HP];
  __shared__ float C2s[B * HP];
  __shared__ float H2f[B * HP];             // h2 f32 for projection
  __shared__ float bias0s[GP];
  __shared__ float bias1s[GP];
  __shared__ float wlinS[HP];

  const int n    = blockIdx.x;
  const int tid  = threadIdx.x;
  const int lane = tid & 31;
  const int wave = tid >> 5;

  const unsigned short* W0ih = Wih0 + (size_t)n * GP * KX;
  const unsigned short* W0hh = Whh0 + (size_t)n * GP * KH;
  const unsigned short* W1ih = Wih1 + (size_t)n * GP * KH;
  const unsigned short* W1hh = Whh1 + (size_t)n * GP * KH;

  // ---- init: zero recurrent state, stage biases / w_lin, warm L2 ----
  for (int i = tid; i < B * KH; i += 256) { H1b[i] = 0; H2b[i] = 0; }
  for (int i = tid; i < B * HP; i += 256) { C1s[i] = 0.0f; C2s[i] = 0.0f; }
  for (int i = tid; i < GP; i += 256) {
    int gate = i / HP, h = i % HP;
    float v0 = 0.0f, v1 = 0.0f;
    if (h < H) {
      int bi = n * G4H + gate * H + h;
      v0 = b_ih0[bi] + b_hh0[bi];
      v1 = b_ih1[bi] + b_hh1[bi];
    }
    bias0s[i] = v0;
    bias1s[i] = v1;
  }
  for (int i = tid; i < HP; i += 256) wlinS[i] = (i < H) ? w_lin[i] : 0.0f;
  // Prefetch the node's layer-1 weights (used a few microseconds from now).
  for (size_t off = (size_t)tid * 128; off < (size_t)GP * KH; off += 256u * 128u) {
    __builtin_prefetch(W1ih + off, 0, 1);
    __builtin_prefetch(W1hh + off, 0, 1);
  }
  __syncthreads();

  const float blin = b_lin[0];
  float accReg = 0.0f;  // running sum_t w_end[t] * (h2 . w_lin + b_lin)

  for (int t = 0; t < T; ++t) {
    // ---- stage x_t as bf16 [B][KX] ----
    for (int e = tid; e < B * KX; e += 256) {
      int b = e >> 5, f = e & 31;
      float v = 0.0f;
      if (f < F) v = x[(((size_t)b * T + t) * NN + n) * F + f];
      Xb[e] = f2bf(v);
    }
    __syncthreads();

    // ---- layer 0 GEMM: Gs = Xb @ W0ih^T + H1b @ W0hh^T ----
    for (int tile = wave; tile < (B / 16) * (GP / 16); tile += NW) {
      int mt = (tile & 3) << 4;
      int nt = (tile >> 2) << 4;
      v8f acc = {};
      v16bf a = load_a_frag(Xb, mt, 0, KX, lane);
      v16bf bm = load_b_frag(W0ih, nt, 0, KX, lane);
      acc = wmma_bf16(a, bm, acc);
#pragma unroll
      for (int kk = 0; kk < KH; kk += 32) {
        a  = load_a_frag(H1b, mt, kk, KH, lane);
        bm = load_b_frag(W0hh, nt, kk, KH, lane);
        acc = wmma_bf16(a, bm, acc);
      }
      store_c_frag(Gs, mt, nt, lane, acc);
    }
    __syncthreads();

    // ---- layer 0 cell update ----
    for (int e = tid; e < B * H; e += 256) {
      int b = e / H, h = e - b * H;
      const float* gr = Gs + (size_t)b * GP;
      float iv = sigf(gr[h]            + bias0s[h]);
      float fv = sigf(gr[HP + h]       + bias0s[HP + h]);
      float gv = tanhf(gr[2 * HP + h]  + bias0s[2 * HP + h]);
      float ov = sigf(gr[3 * HP + h]   + bias0s[3 * HP + h]);
      float c  = fv * C1s[b * HP + h] + iv * gv;
      float hv = ov * tanhf(c);
      C1s[b * HP + h] = c;
      H1b[b * KH + h] = f2bf(hv);
      if (t == T - 1) {
        hn[((size_t)(n * 2 + 0) * B + b) * H + h] = hv;
        cn[((size_t)(n * 2 + 0) * B + b) * H + h] = c;
      }
    }
    __syncthreads();

    // ---- layer 1 GEMM: Gs = H1b @ W1ih^T + H2b @ W1hh^T ----
    for (int tile = wave; tile < (B / 16) * (GP / 16); tile += NW) {
      int mt = (tile & 3) << 4;
      int nt = (tile >> 2) << 4;
      v8f acc = {};
#pragma unroll
      for (int kk = 0; kk < KH; kk += 32) {
        v16bf a  = load_a_frag(H1b, mt, kk, KH, lane);
        v16bf bm = load_b_frag(W1ih, nt, kk, KH, lane);
        acc = wmma_bf16(a, bm, acc);
        a  = load_a_frag(H2b, mt, kk, KH, lane);
        bm = load_b_frag(W1hh, nt, kk, KH, lane);
        acc = wmma_bf16(a, bm, acc);
      }
      store_c_frag(Gs, mt, nt, lane, acc);
    }
    __syncthreads();

    // ---- layer 1 cell update ----
    for (int e = tid; e < B * H; e += 256) {
      int b = e / H, h = e - b * H;
      const float* gr = Gs + (size_t)b * GP;
      float iv = sigf(gr[h]            + bias1s[h]);
      float fv = sigf(gr[HP + h]       + bias1s[HP + h]);
      float gv = tanhf(gr[2 * HP + h]  + bias1s[2 * HP + h]);
      float ov = sigf(gr[3 * HP + h]   + bias1s[3 * HP + h]);
      float c  = fv * C2s[b * HP + h] + iv * gv;
      float hv = ov * tanhf(c);
      C2s[b * HP + h] = c;
      H2b[b * KH + h] = f2bf(hv);
      H2f[b * HP + h] = hv;
      if (t == T - 1) {
        hn[((size_t)(n * 2 + 1) * B + b) * H + h] = hv;
        cn[((size_t)(n * 2 + 1) * B + b) * H + h] = c;
      }
    }
    __syncthreads();

    // ---- projection + conv-over-T accumulation (one thread per batch) ----
    if (tid < B) {
      float dot = 0.0f;
#pragma unroll 4
      for (int h = 0; h < H; ++h) dot += H2f[tid * HP + h] * wlinS[h];
      accReg += w_end[t] * (dot + blin);
    }
    __syncthreads();
  }

  if (tid < B) out[(size_t)tid * NN + n] = accReg + b_end[0];
}

// ---------- launch ----------

extern "C" void kernel_launch(void* const* d_in, const int* in_sizes, int n_in,
                              void* d_out, int out_size, void* d_ws, size_t ws_size,
                              hipStream_t stream) {
  (void)in_sizes; (void)n_in; (void)out_size; (void)ws_size;

  const float* x     = (const float*)d_in[0];
  const float* W_ih0 = (const float*)d_in[1];
  const float* W_hh0 = (const float*)d_in[2];
  const float* b_ih0 = (const float*)d_in[3];
  const float* b_hh0 = (const float*)d_in[4];
  const float* W_ih1 = (const float*)d_in[5];
  const float* W_hh1 = (const float*)d_in[6];
  const float* b_ih1 = (const float*)d_in[7];
  const float* b_hh1 = (const float*)d_in[8];
  const float* w_lin = (const float*)d_in[9];
  const float* b_lin = (const float*)d_in[10];
  const float* w_end = (const float*)d_in[11];
  const float* b_end = (const float*)d_in[12];

  const size_t S0 = (size_t)NN * GP * KX;  // ih0 (bf16 elements)
  const size_t S1 = (size_t)NN * GP * KH;  // hh0 / ih1 / hh1
  unsigned short* ws0 = (unsigned short*)d_ws;
  unsigned short* ws1 = ws0 + S0;
  unsigned short* ws2 = ws1 + S1;
  unsigned short* ws3 = ws2 + S1;

  const int tot0 = NN * GP * KX;
  const int tot1 = NN * GP * KH;
  int blk0 = (tot0 + 255) / 256;
  int blk1 = (tot1 + 255) / 256;
  convert_w_kernel<<<blk0, 256, 0, stream>>>(W_ih0, ws0, F, KX, tot0);
  convert_w_kernel<<<blk1, 256, 0, stream>>>(W_hh0, ws1, H, KH, tot1);
  convert_w_kernel<<<blk1, 256, 0, stream>>>(W_ih1, ws2, H, KH, tot1);
  convert_w_kernel<<<blk1, 256, 0, stream>>>(W_hh1, ws3, H, KH, tot1);

  float* out = (float*)d_out;
  float* hnp = out + (size_t)B * NN;
  float* cnp = hnp + (size_t)NN * 2 * B * H;

  airspace_lstm_kernel<<<NN, 256, 0, stream>>>(
      x, b_ih0, b_hh0, b_ih1, b_hh1, w_lin, b_lin, w_end, b_end,
      ws0, ws1, ws2, ws3, out, hnp, cnp);
}
